// GNNAgent_55052890800739
// MI455X (gfx1250) — matrix operand order repaired and checked by reading, not compiled, not run on hardware
//
#include <hip/hip_runtime.h>
#include <cstdint>
#include <cstddef>

#define HID   128
#define HEADS 2
#define DHEAD 64
#define NKW   100000
#define NDOC  50000
#define NEDGE 1000000

typedef __attribute__((ext_vector_type(16))) __bf16 v16bf;
typedef __attribute__((ext_vector_type(8)))  float  v8f;

union FragU { v16bf v; uint4 q[2]; };

// fragment-major index for the B (weight) matrix: tile (kc,nt), lane, element
// lane L holds col = nt*16 + (L&15), K = kc*32 + (L>>4)*16 + e  (contiguous k order, same as A)
__device__ __forceinline__ int frag_index(int K, int col) {
  int kc = K >> 5, kr = K & 31;
  int half = kr >> 4, e = kr & 15;
  int nt = col >> 4, l16 = col & 15;
  return (((kc << 3) + nt) * 32 + (half << 4) + l16) * 16 + e;
}

// ---------------------------------------------------------------- utilities
__device__ __forceinline__ void atomicMaxF(float* addr, float val) {
  if (val >= 0.0f) atomicMax((int*)addr, __float_as_int(val));
  else             atomicMin((unsigned int*)addr, __float_as_uint(val));
}

// f32 -> bf16 (row-major, for activations), optional exact GELU first
__global__ void cvt_bf16_kernel(const float* __restrict__ in, __bf16* __restrict__ out,
                                long n, int gelu) {
  long i = (long)blockIdx.x * blockDim.x + threadIdx.x;
  if (i >= n) return;
  float x = in[i];
  if (gelu) x = 0.5f * x * (1.0f + erff(x * 0.70710678118654752f));
  out[i] = (__bf16)x;
}

// f32 row-major weight [K=128, N=128] -> bf16 fragment-major
__global__ void cvt_weight_kernel(const float* __restrict__ W, __bf16* __restrict__ out) {
  int idx = blockIdx.x * blockDim.x + threadIdx.x;
  if (idx >= HID * HID) return;
  int K = idx >> 7, col = idx & 127;
  out[frag_index(K, col)] = (__bf16)W[idx];
}

__global__ void fill_kernel(float* __restrict__ p, long n, float v) {
  long i = (long)blockIdx.x * blockDim.x + threadIdx.x;
  if (i < n) p[i] = v;
}

__global__ void mul_kernel(float* __restrict__ a, const float* __restrict__ b, long n) {
  long i = (long)blockIdx.x * blockDim.x + threadIdx.x;
  if (i < n) a[i] *= b[i];
}

__global__ void copy_kernel(const float* __restrict__ a, float* __restrict__ b, long n) {
  long i = (long)blockIdx.x * blockDim.x + threadIdx.x;
  if (i < n) b[i] = a[i];
}

// fold per-head rel matrix into projection weight (output fragment-major bf16):
// Wout[i, h*64+e] = sum_d W[i, h*64+d] * rel[h,d,e];  bout likewise from b
__global__ void compose_rel_kernel(const float* __restrict__ W, const float* __restrict__ b,
                                   const float* __restrict__ rel,
                                   __bf16* __restrict__ Wout, float* __restrict__ bout) {
  int idx = blockIdx.x * blockDim.x + threadIdx.x;      // HID*HID threads
  if (idx >= HID * HID) return;
  int i = idx >> 7;
  int c = idx & 127;
  int h = c >> 6, e = c & 63;
  const float* wr = W + (long)i * HID + h * DHEAD;
  const float* rr = rel + (long)h * DHEAD * DHEAD + e;
  float acc = 0.f;
#pragma unroll 8
  for (int d = 0; d < DHEAD; ++d) acc += wr[d] * rr[(long)d * DHEAD];
  Wout[frag_index(i, c)] = (__bf16)acc;
  if (i == 0) {
    float bb = 0.f;
    for (int d = 0; d < DHEAD; ++d) bb += b[h * DHEAD + d] * rr[(long)d * DHEAD];
    bout[c] = bb;
  }
}

// ------------------------------------------------- WMMA GEMM  C = A@B + bias
// A: [M,128] bf16 row-major; Bsw: 128x128 bf16 FRAGMENT-MAJOR in global;
// C: [M,128] f32; act: 0 none, 1 relu. 8 waves/block, one 16-row strip each.
__global__ void __launch_bounds__(256)
gemm128_bf16(const __bf16* __restrict__ A, const __bf16* __restrict__ Bsw,
             const float* __restrict__ bias, float* __restrict__ C,
             int M, int act) {
  __shared__ __bf16 Bs[HID * HID];                       // 32 KB, fragment-major
  {
    // async DMA the weight tile into LDS (ASYNCcnt path), 16B per lane per op
    unsigned long gbase = (unsigned long)(size_t)Bsw;
    unsigned      lbase = (unsigned)(size_t)&Bs[0];      // low 32b = LDS offset
    for (int i = threadIdx.x; i < (HID * HID) / 8; i += blockDim.x) {
      unsigned      laddr = lbase + (unsigned)i * 16u;
      unsigned long gaddr = gbase + (unsigned long)i * 16u;
      asm volatile("global_load_async_to_lds_b128 %0, %1, off"
                   :: "v"(laddr), "v"(gaddr) : "memory");
    }
    asm volatile("s_wait_asynccnt 0" ::: "memory");
  }
  __syncthreads();

  int wid    = threadIdx.x >> 5;
  int lane   = threadIdx.x & 31;
  int strip  = blockIdx.x * 8 + wid;                     // 16 rows per strip
  int strips = M >> 4;
  if (strip >= strips) return;

  int half = lane >> 4;
  int l16  = lane & 15;
  int rowBase = strip * 16;
  int aRow = rowBase + l16;
  const __bf16* arow = A + (size_t)aRow * HID;

  v8f acc[8] = {};
#pragma unroll
  for (int kc = 0; kc < 4; ++kc) {
    int kb = kc * 32;
    // A fragment: two contiguous 16B spans per lane (contiguous k order, same as B)
    FragU au;
    const uint4* ap = (const uint4*)(arow + kb + half * 16);
    au.q[0] = ap[0];                                     // K = kb+half*16 .. +7
    au.q[1] = ap[1];                                     // K = kb+half*16+8 .. +15
    if (kc < 3) __builtin_prefetch(arow + kb + 32, 0, 1);
#pragma unroll
    for (int nt = 0; nt < 8; ++nt) {
      FragU bu;
      const uint4* bt = (const uint4*)(Bs + (((kc << 3) + nt) * 32 + lane) * 16);
      bu.q[0] = bt[0];
      bu.q[1] = bt[1];
      acc[nt] = __builtin_amdgcn_wmma_f32_16x16x32_bf16(
          false, au.v, false, bu.v, (short)0, acc[nt], false, false);
    }
  }
  // store: VGPR r -> row r (lanes 0-15) / r+8 (lanes 16-31), col = lane%16
#pragma unroll
  for (int nt = 0; nt < 8; ++nt) {
    int col = nt * 16 + l16;
    float bv = bias[col];
#pragma unroll
    for (int r = 0; r < 8; ++r) {
      int row = rowBase + r + half * 8;
      float v = acc[nt][r] + bv;
      if (act == 1) v = fmaxf(v, 0.f);
      C[(size_t)row * HID + col] = v;
    }
  }
}

// ------------------------------------------------- feedback first layer (K=3)
__global__ void fb1_kernel(const float* __restrict__ xfb, const float* __restrict__ W1,
                           const float* __restrict__ b1, __bf16* __restrict__ out) {
  int idx = blockIdx.x * blockDim.x + threadIdx.x;
  if (idx >= NDOC * HID) return;
  int n = idx >> 7, c = idx & 127;
  float acc = b1[c] + xfb[n * 3 + 0] * W1[0 * HID + c]
                    + xfb[n * 3 + 1] * W1[1 * HID + c]
                    + xfb[n * 3 + 2] * W1[2 * HID + c];
  out[idx] = (__bf16)fmaxf(acc, 0.f);
}

// ------------------------------------------------- edge kernels
__global__ void edge_alpha_kernel(const int* __restrict__ src, const int* __restrict__ dst,
                                  const float* __restrict__ kP, const float* __restrict__ qP,
                                  const float* __restrict__ prel,
                                  float* __restrict__ alpha, float* __restrict__ amax) {
  int t = blockIdx.x * blockDim.x + threadIdx.x;
  if (t >= NEDGE * 2) return;
  int e = t >> 1, h = t & 1;
  int s = src[e], d = dst[e];
  const float* kr = kP + (long)s * HID + h * DHEAD;
  const float* qr = qP + (long)d * HID + h * DHEAD;
  float acc = 0.f;
#pragma unroll 8
  for (int i = 0; i < DHEAD; ++i) acc += kr[i] * qr[i];
  acc *= prel[h] * 0.125f;                               // 1/sqrt(64)
  alpha[t] = acc;
  atomicMaxF(&amax[(long)d * 2 + h], acc);
}

__global__ void amax_fix_kernel(float* __restrict__ amax, long n) {
  long i = (long)blockIdx.x * blockDim.x + threadIdx.x;
  if (i < n && amax[i] < -3.0e38f) amax[i] = 0.f;        // where(isfinite, amax, 0)
}

__global__ void edge_exp_kernel(const int* __restrict__ dst, float* __restrict__ alpha,
                                const float* __restrict__ amax, float* __restrict__ denom) {
  int t = blockIdx.x * blockDim.x + threadIdx.x;
  if (t >= NEDGE * 2) return;
  int e = t >> 1, h = t & 1;
  int d = dst[e];
  float ex = expf(alpha[t] - amax[(long)d * 2 + h]);
  alpha[t] = ex;
  atomicAdd(&denom[(long)d * 2 + h], ex);
}

__global__ void edge_scatter_kernel(const int* __restrict__ src, const int* __restrict__ dst,
                                    const float* __restrict__ ex, const float* __restrict__ denom,
                                    const float* __restrict__ vP, float* __restrict__ agg) {
  long t = (long)blockIdx.x * blockDim.x + threadIdx.x;
  if (t >= (long)NEDGE * HID) return;
  int e = (int)(t >> 7);
  int c = (int)(t & 127);
  int h = c >> 6;
  int s = src[e], d = dst[e];
  float attn = ex[(long)e * 2 + h] / (denom[(long)d * 2 + h] + 1e-16f);
  atomicAdd(&agg[(long)d * HID + c], vP[(long)s * HID + c] * attn);
}

__global__ void skip_combine_kernel(const float* __restrict__ woOut, float* __restrict__ dstF,
                                    const float* __restrict__ skip, long n, int act) {
  long i = (long)blockIdx.x * blockDim.x + threadIdx.x;
  if (i >= n) return;
  float a = 1.f / (1.f + expf(-skip[0]));
  float v = a * woOut[i] + (1.f - a) * dstF[i];
  if (act) v = fmaxf(v, 0.f);
  dstF[i] = v;
}

__global__ void final_dot_kernel(const float* __restrict__ docF, const float* __restrict__ W,
                                 const float* __restrict__ b, float* __restrict__ out) {
  int n = blockIdx.x * blockDim.x + threadIdx.x;
  if (n >= NDOC) return;
  const float* r = docF + (long)n * HID;
  float acc = b[0];
#pragma unroll 8
  for (int c = 0; c < HID; ++c) acc += r[c] * W[c];
  out[n] = acc;
}

// ---------------------------------------------------------------- launch
extern "C" void kernel_launch(void* const* d_in, const int* in_sizes, int n_in,
                              void* d_out, int out_size, void* d_ws, size_t ws_size,
                              hipStream_t stream) {
  (void)in_sizes; (void)n_in; (void)out_size; (void)ws_size;
  auto F = [&](int i) { return (const float*)d_in[i]; };
  auto I = [&](int i) { return (const int*)d_in[i]; };

  // workspace carve (256B aligned)
  char* wsb = (char*)d_ws;
  size_t off = 0;
  auto carve = [&](size_t bytes) -> void* {
    void* p = wsb + off; off += (bytes + 255) & ~(size_t)255; return p;
  };
  float*  kwF    = (float*)carve((size_t)NKW  * HID * 4);
  float*  docF   = (float*)carve((size_t)NDOC * HID * 4);
  float*  fF     = (float*)carve((size_t)NDOC * HID * 4);
  float*  p0     = (float*)carve((size_t)NKW  * HID * 4);   // kP, later Wo out
  float*  p1     = (float*)carve((size_t)NKW  * HID * 4);   // vP
  float*  p2     = (float*)carve((size_t)NKW  * HID * 4);   // qP
  float*  p3     = (float*)carve((size_t)NKW  * HID * 4);   // agg
  float*  alphaB = (float*)carve((size_t)NEDGE * 2 * 4);
  float*  amaxB  = (float*)carve((size_t)NKW * 2 * 4);
  float*  denomB = (float*)carve((size_t)NKW * 2 * 4);
  __bf16* hA     = (__bf16*)carve((size_t)NKW * HID * 2);   // src feats / gelu(agg)
  __bf16* hB     = (__bf16*)carve((size_t)NKW * HID * 2);   // dst feats / fb hidden
  __bf16* WkH    = (__bf16*)carve((size_t)HID * HID * 2);
  __bf16* WvH    = (__bf16*)carve((size_t)HID * HID * 2);
  __bf16* WqH    = (__bf16*)carve((size_t)HID * HID * 2);
  __bf16* WoH    = (__bf16*)carve((size_t)HID * HID * 2);
  float*  bkC    = (float*)carve(HID * 4);
  float*  bvC    = (float*)carve(HID * 4);

  const int TPB = 256;
  auto g = [](long n) { return (unsigned)((n + 255) / 256); };
  auto gemm = [&](const __bf16* A, const __bf16* B, const float* bias,
                  float* C, int M, int act) {
    unsigned blocks = (unsigned)(((M >> 4) + 7) / 8);
    gemm128_bf16<<<blocks, 256, 0, stream>>>(A, B, bias, C, M, act);
  };
  auto cvt = [&](const float* in, __bf16* out, long n, int gelu) {
    cvt_bf16_kernel<<<g(n), TPB, 0, stream>>>(in, out, n, gelu);
  };
  auto cvtW = [&](const float* W, __bf16* out) {
    cvt_weight_kernel<<<64, 256, 0, stream>>>(W, out);
  };

  // ---- dense0: kw = relu(x_kw@Wkw+b), doc = relu(x_doc@Wdoc+b)
  cvt(F(0), hA, (long)NKW * HID, 0);
  cvtW(F(6), WkH);
  gemm(hA, WkH, F(7), kwF, NKW, 1);
  cvt(F(1), hB, (long)NDOC * HID, 0);
  cvtW(F(8), WvH);
  gemm(hB, WvH, F(9), docF, NDOC, 1);

  // ---- feedback MLP: f = relu(xfb@W1+b1)@W2+b2
  fb1_kernel<<<g((long)NDOC * HID), TPB, 0, stream>>>(F(2), F(10), F(11), hB);
  cvtW(F(12), WqH);
  gemm(hB, WqH, F(13), fF, NDOC, 0);

  // ---- one HGT layer
  auto hgt = [&](const float* srcF, int n_src, float* dstF, int n_dst,
                 const int* eptr, int pb, int act) {
    const float *Wk = F(pb + 0), *bk = F(pb + 1), *Wq = F(pb + 2), *bq = F(pb + 3);
    const float *Wv = F(pb + 4), *bv = F(pb + 5), *arel = F(pb + 6), *mrel = F(pb + 7);
    const float *prel = F(pb + 8), *Wo = F(pb + 9), *bo = F(pb + 10), *skp = F(pb + 11);
    compose_rel_kernel<<<64, 256, 0, stream>>>(Wk, bk, arel, WkH, bkC);
    compose_rel_kernel<<<64, 256, 0, stream>>>(Wv, bv, mrel, WvH, bvC);
    cvtW(Wq, WqH);
    cvtW(Wo, WoH);
    cvt(srcF, hA, (long)n_src * HID, 0);
    cvt(dstF, hB, (long)n_dst * HID, 0);
    gemm(hA, WkH, bkC, p0, n_src, 0);       // k' = src @ (Wk·a_rel)
    gemm(hA, WvH, bvC, p1, n_src, 0);       // v' = src @ (Wv·m_rel)
    gemm(hB, WqH, bq,  p2, n_dst, 0);       // q  = dst @ Wq
    fill_kernel<<<g((long)n_dst * 2), TPB, 0, stream>>>(amaxB, (long)n_dst * 2,
                                                        -__builtin_inff());
    fill_kernel<<<g((long)n_dst * 2), TPB, 0, stream>>>(denomB, (long)n_dst * 2, 0.f);
    fill_kernel<<<g((long)n_dst * HID), TPB, 0, stream>>>(p3, (long)n_dst * HID, 0.f);
    const int* src = eptr;
    const int* dst = eptr + NEDGE;
    edge_alpha_kernel<<<g(2L * NEDGE), TPB, 0, stream>>>(src, dst, p0, p2, prel,
                                                         alphaB, amaxB);
    amax_fix_kernel<<<g((long)n_dst * 2), TPB, 0, stream>>>(amaxB, (long)n_dst * 2);
    edge_exp_kernel<<<g(2L * NEDGE), TPB, 0, stream>>>(dst, alphaB, amaxB, denomB);
    edge_scatter_kernel<<<g((long)NEDGE * HID), TPB, 0, stream>>>(src, dst, alphaB,
                                                                  denomB, p1, p3);
    cvt(p3, hA, (long)n_dst * HID, 1);      // exact gelu -> bf16
    gemm(hA, WoH, bo, p0, n_dst, 0);        // gelu(agg) @ Wo + bo
    skip_combine_kernel<<<g((long)n_dst * HID), TPB, 0, stream>>>(p0, dstF, skp,
                                                                  (long)n_dst * HID, act);
  };

  // layer1: kw -> doc (relu), then doc *= f
  hgt(kwF, NKW, docF, NDOC, I(3), 14, 1);
  mul_kernel<<<g((long)NDOC * HID), TPB, 0, stream>>>(docF, fF, (long)NDOC * HID);
  // layer2: doc -> kw (relu)
  hgt(docF, NDOC, kwF, NKW, I(4), 26, 1);
  // layer3: kw -> doc (no act)
  hgt(kwF, NKW, docF, NDOC, I(5), 38, 0);

  // final head + outputs (tuple: out[NDOC,1] then kw[NKW,HID])
  float* out = (float*)d_out;
  final_dot_kernel<<<g(NDOC), TPB, 0, stream>>>(docF, F(50), F(51), out);
  copy_kernel<<<g((long)NKW * HID), TPB, 0, stream>>>(kwF, out + NDOC, (long)NKW * HID);
}
